// MultiHeadAttention_38946763440365
// MI455X (gfx1250) — matrix operand order repaired and checked by reading, not compiled, not run on hardware
//
#include <hip/hip_runtime.h>
#include <hip/hip_bf16.h>

// ---------------------------------------------------------------------------
// MHA block for MI455X (gfx1250, wave32, WMMA f32_16x16x32_f16 + TDM)
//  B=4, S=2048, D=1024, H=16, DQ=DV=64
// ---------------------------------------------------------------------------

#define BB 4
#define SS 2048
#define DD 1024
#define HH 16
#define DH 64   // head dim (DQ == DV)

typedef __attribute__((ext_vector_type(16))) _Float16 v16h;
typedef __attribute__((ext_vector_type(8)))  float    v8f;
typedef __attribute__((ext_vector_type(4)))  unsigned int u32x4;
typedef __attribute__((ext_vector_type(8)))  int      i32x8;
typedef __attribute__((ext_vector_type(4)))  int      i32x4;

union Frag16 { v16h h; unsigned int u[8]; };

// A-fragment (16x32, f16, row-major source, leading dim lda):
// lane m = lane&15, half hh = lane>>4; VGPR v holds K pair
//   k = (v/4)*16 + hh*8 + (v%4)*2   (ISA 7.12.2, 16-bit A 16x32)
__device__ __forceinline__ v16h load_a_frag(const _Float16* A, int lda,
                                            int m_base, int k_base) {
  const int lane = threadIdx.x & 31;
  const int m = lane & 15, hh = lane >> 4;
  const _Float16* row = A + (size_t)(m_base + m) * lda + k_base;
  Frag16 f;
#pragma unroll
  for (int v = 0; v < 8; ++v) {
    const int off = ((v >> 2) << 4) + (hh << 3) + ((v & 3) << 1);
    f.u[v] = *reinterpret_cast<const unsigned int*>(row + off);
  }
  return f.h;
}

// B-fragment (32x16, f16) from a K-contiguous LDS tile: tile[n][0..31] halves,
// row stride 32. lane n = lane&15, hh = lane>>4; VGPR v = pair k = (v/4)*16+hh*8+(v%4)*2.
__device__ __forceinline__ v16h load_b_frag_lds(const _Float16* tile,
                                                int n_base) {
  const int lane = threadIdx.x & 31;
  const int n = lane & 15, hh = lane >> 4;
  const _Float16* p = tile + (n_base + n) * 32 + (hh << 3);
  Frag16 f;
#pragma unroll
  for (int v = 0; v < 8; ++v)
    f.u[v] = *reinterpret_cast<const unsigned int*>(p + ((v >> 2) << 4) + ((v & 3) << 1));
  return f.h;
}

// B-fragment where the per-lane row pointer is precomputed (per-lane column).
__device__ __forceinline__ v16h load_b_frag_lane(const _Float16* lane_row,
                                                 int k_base) {
  const int hh = (threadIdx.x >> 4) & 1;
  const _Float16* p = lane_row + k_base + (hh << 3);
  Frag16 f;
#pragma unroll
  for (int v = 0; v < 8; ++v)
    f.u[v] = *reinterpret_cast<const unsigned int*>(p + ((v >> 2) << 4) + ((v & 3) << 1));
  return f.h;
}

// ---------------------------------------------------------------------------
// TDM: DMA a 2D f16 tile [tile_n rows x 32 halves] from global (row stride
// src_ld halves) into LDS at lds_off, contiguous row-major. D# per ISA ch.8:
// group0: count=1, lds_addr, global_addr[56:0], type=2
// group1: data_size=2B, tensor dims == tile dims (no OOB), dim0 stride.
// amdgpu-toolchain (clang-23) 6-arg builtin form.
// ---------------------------------------------------------------------------
__device__ __forceinline__ void tdm_load_tile(unsigned lds_off,
                                              const _Float16* gptr,
                                              unsigned tile_n,
                                              unsigned src_ld) {
  const unsigned long long ga = (unsigned long long)gptr;
  u32x4 g0;
  g0[0] = 1u;                                        // count=1, user mode
  g0[1] = lds_off;                                   // lds_addr (bytes)
  g0[2] = (unsigned)(ga & 0xffffffffull);            // global_addr[31:0]
  g0[3] = (unsigned)((ga >> 32) & 0x01ffffffull)     // global_addr[56:32]
          | 0x80000000u;                             // type=2 ("image")
  const unsigned tk = 32u;                           // tile/tensor dim0
  i32x8 g1;
  g1[0] = (int)(1u << 16);                           // data_size=1 -> 2 bytes
  g1[1] = (int)((tk & 0xffffu) << 16);               // tensor_dim0[15:0]
  g1[2] = (int)(((tk >> 16) & 0xffffu) |
                ((tile_n & 0xffffu) << 16));         // dim0 hi | tensor_dim1 lo
  g1[3] = (int)(((tile_n >> 16) & 0xffffu) |
                ((tk & 0xffffu) << 16));             // dim1 hi | tile_dim0
  g1[4] = (int)(tile_n & 0xffffu);                   // tile_dim1 (tile_dim2=0)
  g1[5] = (int)src_ld;                               // tensor_dim0_stride lo
  g1[6] = 0;                                         // stride hi | dim1_stride
  g1[7] = 0;
  const i32x4 gz4 = {0, 0, 0, 0};                    // groups 2/3 unused (2D)
  const i32x8 gz8 = {0, 0, 0, 0, 0, 0, 0, 0};
  __builtin_amdgcn_tensor_load_to_lds(g0, g1, gz4, gz4, gz8, 0);
}

// ---------------------------------------------------------------------------
// 1) LayerNorm: one 256-thread block per row of D=1024.
// ---------------------------------------------------------------------------
__global__ void ln_kernel(const float* __restrict__ x,
                          const float* __restrict__ gamma,
                          const float* __restrict__ beta,
                          float* __restrict__ xn,
                          _Float16* __restrict__ xnh) {
  const int row = blockIdx.x;
  const float* xr = x + (size_t)row * DD;
  __shared__ float red[256];
  const int tid = threadIdx.x;

  float s = 0.f;
  for (int i = tid; i < DD; i += 256) s += xr[i];
  red[tid] = s; __syncthreads();
  for (int o = 128; o > 0; o >>= 1) {
    if (tid < o) red[tid] += red[tid + o];
    __syncthreads();
  }
  const float mu = red[0] * (1.f / DD);
  __syncthreads();

  float vs = 0.f;
  for (int i = tid; i < DD; i += 256) { float d = xr[i] - mu; vs += d * d; }
  red[tid] = vs; __syncthreads();
  for (int o = 128; o > 0; o >>= 1) {
    if (tid < o) red[tid] += red[tid + o];
    __syncthreads();
  }
  const float rstd = rsqrtf(red[0] * (1.f / DD) + 1e-3f);

  for (int i = tid; i < DD; i += 256) {
    const float y = (xr[i] - mu) * rstd * gamma[i] + beta[i];
    xn[(size_t)row * DD + i]  = y;
    xnh[(size_t)row * DD + i] = (_Float16)y;
  }
}

// ---------------------------------------------------------------------------
// 2) Weight transpose + downconvert: W[K,N] f32 -> WT[N,K] f16
// ---------------------------------------------------------------------------
__global__ void wt_kernel(const float* __restrict__ W, _Float16* __restrict__ WT,
                          int Kdim, int Ndim) {
  const int idx = blockIdx.x * blockDim.x + threadIdx.x;
  if (idx >= Kdim * Ndim) return;
  const int k = idx / Ndim, n = idx % Ndim;
  WT[(size_t)n * Kdim + k] = (_Float16)W[idx];
}

// ---------------------------------------------------------------------------
// 3) Fused QKV projection GEMM, TDM-staged weights.
//    Block = 128(M) x 64(N); 8 waves each own a 16x64 slice and share the
//    double-buffered 64x32 f16 weight tile DMA'd into LDS by wave 0.
//    M = B*S = 8192, N = 3072 (Q | K | V).
// ---------------------------------------------------------------------------
__global__ void gemm_qkv_kernel(const _Float16* __restrict__ A,   // [8192,1024]
                                const _Float16* __restrict__ WqT, // [1024,1024]
                                const _Float16* __restrict__ WkvT,// [2048,1024]
                                const float* __restrict__ b_q,
                                const float* __restrict__ b_kv,
                                _Float16* __restrict__ Qf,  // [B*S, H*DH]
                                _Float16* __restrict__ Kf,  // [B*S, H*DH]
                                _Float16* __restrict__ Vt)  // [B,H,DH,S]
{
  __shared__ _Float16 btile[2][64 * 32];   // 2 x 4KB double buffer

  const int nb = blockIdx.x % 48;          // 48 N-blocks of 64 over 3072
  const int mb = blockIdx.x / 48;          // 64 M-blocks of 128 over 8192
  const int n0 = nb * 64;
  const int m0 = mb * 128 + (int)(threadIdx.x >> 5) * 16;
  const bool wave0 = (threadIdx.x < 32);

  // whole 64-wide N block lives on one side of the Q|KV split (1024 % 64 == 0)
  const _Float16* BT = (n0 < 1024) ? (WqT + (size_t)n0 * DD)
                                   : (WkvT + (size_t)(n0 - 1024) * DD);
  const unsigned lds0 = (unsigned)(unsigned long long)(void*)&btile[0][0];
  const unsigned lds1 = (unsigned)(unsigned long long)(void*)&btile[1][0];

  if (wave0) tdm_load_tile(lds0, BT, 64, DD);        // prologue: k=0 -> buf0

  v8f acc[4] = {v8f{}, v8f{}, v8f{}, v8f{}};
  for (int kt = 0; kt < DD / 32; ++kt) {
    const int k = kt * 32;
    __syncthreads();                                  // done reading next buf
    if (wave0) {
      if (kt + 1 < DD / 32) {
        tdm_load_tile((kt & 1) ? lds0 : lds1, BT + (k + 32), 64, DD);
        __builtin_amdgcn_s_wait_tensorcnt(1);         // current tile landed
      } else {
        __builtin_amdgcn_s_wait_tensorcnt(0);
      }
    }
    __syncthreads();                                  // tile visible to all

    const v16h a = load_a_frag(A, DD, m0, k);
    __builtin_prefetch(A + (size_t)m0 * DD + k + 64, 0, 1);
    const _Float16* cur = btile[kt & 1];
#pragma unroll
    for (int t = 0; t < 4; ++t) {
      const v16h b = load_b_frag_lds(cur, t * 16);
      acc[t] = __builtin_amdgcn_wmma_f32_16x16x32_f16(
          false, a, false, b, (short)0, acc[t], false, false);
    }
  }

  const int lane = threadIdx.x & 31;
  const int nlo = lane & 15, hh = lane >> 4;
#pragma unroll
  for (int t = 0; t < 4; ++t) {
    const int n = n0 + t * 16 + nlo;
#pragma unroll
    for (int r = 0; r < 8; ++r) {
      const int m = m0 + r + hh * 8;
      float v = acc[t][r];
      if (n < 1024) {                       // Q (1/sqrt(64) folded in)
        v += b_q[n];
        Qf[(size_t)m * 1024 + n] = (_Float16)(v * 0.125f);
      } else {
        const int c = n - 1024;
        v += b_kv[c];
        if (c < 1024) {                     // K
          Kf[(size_t)m * 1024 + c] = (_Float16)v;
        } else {                            // V -> transposed [b,h,dv,s]
          const int c2 = c - 1024;
          const int h = c2 >> 6, dv = c2 & 63;
          const int b = m >> 11;            // m / S
          const int s = m & (SS - 1);
          Vt[(((size_t)(b * HH + h) * DH) + dv) * SS + s] = (_Float16)v;
        }
      }
    }
  }
}

// ---------------------------------------------------------------------------
// 4) Flash attention: one wave per (b, h, 16-query tile).
//    Online softmax; key-padding mask from lengths; P restaged via LDS.
// ---------------------------------------------------------------------------
__global__ void flash_attn_kernel(const _Float16* __restrict__ Qf,
                                  const _Float16* __restrict__ Kf,
                                  const _Float16* __restrict__ Vt,
                                  const int* __restrict__ lengths,
                                  _Float16* __restrict__ ctx) // [B*S, H*DH]
{
  __shared__ _Float16 pShared[8][16 * 32];   // per-wave P tile
  const int wslot = threadIdx.x >> 5;
  const int wid = blockIdx.x * 8 + wslot;
  const int qt = wid & 127;                  // S/16 tiles
  const int h  = (wid >> 7) & (HH - 1);
  const int b  = wid >> 11;
  const int lane = threadIdx.x & 31;
  const int nlo = lane & 15, hh = lane >> 4;

  const int len = lengths[b];

  const _Float16* Qbase = Qf + ((size_t)(b * SS + qt * 16)) * 1024 + h * DH;
  const _Float16* Kbase = Kf + (size_t)b * SS * 1024 + h * DH;     // +j*1024
  const _Float16* Vbase = Vt + (size_t)(b * HH + h) * DH * SS;     // +dv*SS

  const v16h a_lo = load_a_frag(Qbase, 1024, 0, 0);
  const v16h a_hi = load_a_frag(Qbase, 1024, 0, 32);

  v8f acc[4] = {v8f{}, v8f{}, v8f{}, v8f{}};
  float m_i[8], l_i[8];
#pragma unroll
  for (int r = 0; r < 8; ++r) { m_i[r] = -1e30f; l_i[r] = 0.f; }

  _Float16* P = pShared[wslot];

  // C-layout column key indices (column = lane&15 in both lane halves)
  const int j0base = nlo, j1base = 16 + nlo;

  for (int kb = 0; kb < len; kb += 32) {
    const int j0 = kb + j0base, j1 = kb + j1base;
    const int j0c = j0 < (SS - 1) ? j0 : (SS - 1);
    const int j1c = j1 < (SS - 1) ? j1 : (SS - 1);
    const _Float16* k0row = Kbase + (size_t)j0c * 1024;
    const _Float16* k1row = Kbase + (size_t)j1c * 1024;

    // scores: S0 = keys kb..kb+15, S1 = keys kb+16..kb+31
    v8f s0 = {}, s1 = {};
    s0 = __builtin_amdgcn_wmma_f32_16x16x32_f16(false, a_lo, false,
           load_b_frag_lane(k0row, 0),  (short)0, s0, false, false);
    s0 = __builtin_amdgcn_wmma_f32_16x16x32_f16(false, a_hi, false,
           load_b_frag_lane(k0row, 32), (short)0, s0, false, false);
    s1 = __builtin_amdgcn_wmma_f32_16x16x32_f16(false, a_lo, false,
           load_b_frag_lane(k1row, 0),  (short)0, s1, false, false);
    s1 = __builtin_amdgcn_wmma_f32_16x16x32_f16(false, a_hi, false,
           load_b_frag_lane(k1row, 32), (short)0, s1, false, false);

    // key-padding mask (register writes only; EXEC untouched around WMMA)
    const float NEG = -1e30f;
    if (j0 >= len) {
#pragma unroll
      for (int r = 0; r < 8; ++r) s0[r] = NEG;
    }
    if (j1 >= len) {
#pragma unroll
      for (int r = 0; r < 8; ++r) s1[r] = NEG;
    }

    // online softmax, row stats reduced over each 16-lane half
#pragma unroll
    for (int r = 0; r < 8; ++r) {
      float mx = fmaxf(s0[r], s1[r]);
      mx = fmaxf(mx, __shfl_xor(mx, 1, 16));
      mx = fmaxf(mx, __shfl_xor(mx, 2, 16));
      mx = fmaxf(mx, __shfl_xor(mx, 4, 16));
      mx = fmaxf(mx, __shfl_xor(mx, 8, 16));

      const float mn = fmaxf(m_i[r], mx);
      const float sc = __expf(m_i[r] - mn);
      m_i[r] = mn;
      l_i[r] *= sc;
#pragma unroll
      for (int t = 0; t < 4; ++t) acc[t][r] *= sc;

      const float p0 = __expf(s0[r] - mn);
      const float p1 = __expf(s1[r] - mn);
      float ps = p0 + p1;
      ps += __shfl_xor(ps, 1, 16);
      ps += __shfl_xor(ps, 2, 16);
      ps += __shfl_xor(ps, 4, 16);
      ps += __shfl_xor(ps, 8, 16);
      l_i[r] += ps;

      const int row = r + hh * 8;
      P[row * 32 + nlo]      = (_Float16)p0;
      P[row * 32 + 16 + nlo] = (_Float16)p1;
    }

    // read P back as an A-fragment (16x32) from LDS
    Frag16 fp;
    {
      const int m = nlo;
#pragma unroll
      for (int v = 0; v < 8; ++v) {
        const int off = ((v >> 2) << 4) + (hh << 3) + ((v & 3) << 1);
        fp.u[v] = *reinterpret_cast<const unsigned int*>(&P[m * 32 + off]);
      }
    }

    // ctx += P @ V  (V transposed: rows are dv, contiguous along s)
#pragma unroll
    for (int t = 0; t < 4; ++t) {
      const _Float16* vrow = Vbase + (size_t)(t * 16 + nlo) * SS;
      const v16h bv = load_b_frag_lane(vrow, kb);
      acc[t] = __builtin_amdgcn_wmma_f32_16x16x32_f16(
          false, fp.h, false, bv, (short)0, acc[t], false, false);
    }
  }

  // normalize and store ctx (f16) at [b, q, h*64 + col]
  float inv[8];
#pragma unroll
  for (int r = 0; r < 8; ++r) inv[r] = 1.f / l_i[r];
#pragma unroll
  for (int t = 0; t < 4; ++t) {
#pragma unroll
    for (int r = 0; r < 8; ++r) {
      const int q = qt * 16 + r + hh * 8;
      const int col = h * DH + t * 16 + nlo;
      ctx[((size_t)(b * SS + q)) * 1024 + col] = (_Float16)(acc[t][r] * inv[r]);
    }
  }
}

// ---------------------------------------------------------------------------
// 5) Output projection + bias + residual, TDM-staged weights (f32 out)
// ---------------------------------------------------------------------------
__global__ void gemm_out_kernel(const _Float16* __restrict__ A,   // ctx
                                const _Float16* __restrict__ WoT, // [1024,1024]
                                const float* __restrict__ b_o,
                                const float* __restrict__ resid,  // xn f32
                                float* __restrict__ out)
{
  __shared__ _Float16 btile[2][64 * 32];

  const int nb = blockIdx.x & 15;          // 16 N-blocks of 64 over 1024
  const int mb = blockIdx.x >> 4;          // 64 M-blocks of 128 over 8192
  const int n0 = nb * 64;
  const int m0 = mb * 128 + (int)(threadIdx.x >> 5) * 16;
  const bool wave0 = (threadIdx.x < 32);

  const _Float16* BT = WoT + (size_t)n0 * DD;
  const unsigned lds0 = (unsigned)(unsigned long long)(void*)&btile[0][0];
  const unsigned lds1 = (unsigned)(unsigned long long)(void*)&btile[1][0];

  if (wave0) tdm_load_tile(lds0, BT, 64, DD);

  v8f acc[4] = {v8f{}, v8f{}, v8f{}, v8f{}};
  for (int kt = 0; kt < DD / 32; ++kt) {
    const int k = kt * 32;
    __syncthreads();
    if (wave0) {
      if (kt + 1 < DD / 32) {
        tdm_load_tile((kt & 1) ? lds0 : lds1, BT + (k + 32), 64, DD);
        __builtin_amdgcn_s_wait_tensorcnt(1);
      } else {
        __builtin_amdgcn_s_wait_tensorcnt(0);
      }
    }
    __syncthreads();

    const v16h a = load_a_frag(A, DD, m0, k);
    __builtin_prefetch(A + (size_t)m0 * DD + k + 64, 0, 1);
    const _Float16* cur = btile[kt & 1];
#pragma unroll
    for (int t = 0; t < 4; ++t) {
      const v16h b = load_b_frag_lds(cur, t * 16);
      acc[t] = __builtin_amdgcn_wmma_f32_16x16x32_f16(
          false, a, false, b, (short)0, acc[t], false, false);
    }
  }

  const int lane = threadIdx.x & 31;
  const int nlo = lane & 15, hh = lane >> 4;
#pragma unroll
  for (int t = 0; t < 4; ++t) {
    const int n = n0 + t * 16 + nlo;
#pragma unroll
    for (int r = 0; r < 8; ++r) {
      const int m = m0 + r + hh * 8;
      out[(size_t)m * DD + n] = acc[t][r] + b_o[n] + resid[(size_t)m * DD + n];
    }
  }
}

// ---------------------------------------------------------------------------
// Launch
// ---------------------------------------------------------------------------
extern "C" void kernel_launch(void* const* d_in, const int* in_sizes, int n_in,
                              void* d_out, int out_size, void* d_ws, size_t ws_size,
                              hipStream_t stream) {
  const float* x      = (const float*)d_in[0];   // query_sequences (== kv)
  const int*   lens   = (const int*)  d_in[2];
  const float* W_q    = (const float*)d_in[3];
  const float* b_q    = (const float*)d_in[4];
  const float* W_kv   = (const float*)d_in[5];
  const float* b_kv   = (const float*)d_in[6];
  const float* W_o    = (const float*)d_in[7];
  const float* b_o    = (const float*)d_in[8];
  const float* gamma  = (const float*)d_in[9];
  const float* beta   = (const float*)d_in[10];
  float* out = (float*)d_out;

  const size_t M = (size_t)BB * SS;              // 8192
  char* ws = (char*)d_ws;
  size_t off = 0;
  auto carve = [&](size_t bytes) -> void* {
    void* p = ws + off;
    off += (bytes + 255) & ~(size_t)255;
    return p;
  };
  float*    xn   = (float*)   carve(M * DD * sizeof(float));     // residual
  _Float16* xnh  = (_Float16*)carve(M * DD * sizeof(_Float16));  // LN f16
  _Float16* WqT  = (_Float16*)carve((size_t)1024 * DD * sizeof(_Float16));
  _Float16* WkvT = (_Float16*)carve((size_t)2048 * DD * sizeof(_Float16));
  _Float16* WoT  = (_Float16*)carve((size_t)1024 * DD * sizeof(_Float16));
  _Float16* Qf   = (_Float16*)carve(M * 1024 * sizeof(_Float16));
  _Float16* Kf   = (_Float16*)carve(M * 1024 * sizeof(_Float16));
  _Float16* Vt   = (_Float16*)carve(M * 1024 * sizeof(_Float16));
  _Float16* ctx  = (_Float16*)carve(M * 1024 * sizeof(_Float16) + 4096); // pad

  // 1) LayerNorm (self-attention branch: one LN serves q_norm and kv_used)
  ln_kernel<<<(int)M, 256, 0, stream>>>(x, gamma, beta, xn, xnh);

  // 2) Transpose weights to f16, K-contiguous
  {
    int nq = 1024 * 1024, nkv = 1024 * 2048;
    wt_kernel<<<(nq  + 255) / 256, 256, 0, stream>>>(W_q,  WqT,  1024, 1024);
    wt_kernel<<<(nkv + 255) / 256, 256, 0, stream>>>(W_kv, WkvT, 1024, 2048);
    wt_kernel<<<(nq  + 255) / 256, 256, 0, stream>>>(W_o,  WoT,  1024, 1024);
  }

  // 3) QKV projection: 64 M-blocks x 48 N-blocks, 256 threads (8 waves)
  gemm_qkv_kernel<<<64 * 48, 256, 0, stream>>>(
      xnh, WqT, WkvT, b_q, b_kv, Qf, Kf, Vt);

  // 4) Flash attention: B*H*(S/16) = 8192 waves, 8 per block
  flash_attn_kernel<<<8192 / 8, 256, 0, stream>>>(Qf, Kf, Vt, lens, ctx);

  // 5) Output projection + bias + residual: 64 x 16 blocks
  gemm_out_kernel<<<64 * 16, 256, 0, stream>>>(ctx, WoT, b_o, xn, out);
}
